// ImprovedDiscriminator_46359876993738
// MI455X (gfx1250) — matrix operand order repaired and checked by reading, not compiled, not run on hardware
//
#include <hip/hip_runtime.h>
#include <math.h>

typedef __attribute__((ext_vector_type(16))) __bf16 v16bf;
typedef __attribute__((ext_vector_type(8)))  float  v8f;
typedef __attribute__((ext_vector_type(4)))  float  v4f;

static constexpr int Bb = 128, Tt = 256, Ff = 64, Hh = 512, H4 = 2048;
static constexpr int BT = Bb * Tt;

#define DEVINL __device__ __forceinline__

// ---- WMMA operand loaders: base pointers are precomputed per lane, k offset is a
// compile-time constant after unrolling so it folds into the load immediate field. ----

// A tile (16x32, row-major source): lane base = p + (row0+lanelo)*ld + 16*half.
// VGPRs 0-3 <- base[k0..k0+7], VGPRs 4-7 <- base[k0+8..k0+15].
DEVINL v16bf load_A_at(const __bf16* lane_base, int k0) {
  union { v16bf v; v4f f[2]; } u;
  u.f[0] = *reinterpret_cast<const v4f*>(lane_base + k0);
  u.f[1] = *reinterpret_cast<const v4f*>(lane_base + k0 + 8);
  return u.v;
}

// B tile (32x16, from N-major transposed weight): lane base = p + (n0+lanelo)*ld + 16*half.
// 16 contiguous bf16 = two 16B loads.
DEVINL v16bf load_B_at(const __bf16* lane_base, int k0) {
  union { v16bf v; v4f f[2]; } u;
  u.f[0] = *reinterpret_cast<const v4f*>(lane_base + k0);
  u.f[1] = *reinterpret_cast<const v4f*>(lane_base + k0 + 8);
  return u.v;
}

DEVINL v8f wmma_bf16(v16bf a, v16bf b, v8f c) {
  return __builtin_amdgcn_wmma_f32_16x16x32_bf16(false, a, false, b, (short)0, c, false, false);
}

DEVINL float sigmoid_(float x) { return 1.0f / (1.0f + __expf(-x)); }

// ---- utility kernels ----
__global__ void k_f32_to_bf16(const float* __restrict__ in, __bf16* __restrict__ out, int n) {
  int i = blockIdx.x * blockDim.x + threadIdx.x;
  if (i < n) out[i] = (__bf16)in[i];
}

// out[c*R + r] = (bf16) in[r*C + c]   (store W^T, N-major)
__global__ void k_transpose_bf16(const float* __restrict__ in, __bf16* __restrict__ out, int R, int C) {
  int i = blockIdx.x * blockDim.x + threadIdx.x;
  if (i < R * C) { int r = i / C, c = i % C; out[(size_t)c * R + r] = (__bf16)in[i]; }
}

__global__ void k_zero_state(__bf16* __restrict__ h, float* __restrict__ c, int n) {
  int i = blockIdx.x * blockDim.x + threadIdx.x;
  if (i < n) { h[i] = (__bf16)0.0f; c[i] = 0.0f; }
}

// ---- dense1 + spectral row-norm: h0 = l2norm_rows(x@W1 + b1) * sqrt(H) ----
// grid.x = BT/16 row tiles; block = 128 (4 waves), wave handles 8 N-tiles (512 cols total).
__global__ void k_dense1_norm(const __bf16* __restrict__ xbf, const __bf16* __restrict__ W1t,
                              const float* __restrict__ b1, __bf16* __restrict__ h0) {
  __shared__ float ssq[16];
  int mt = blockIdx.x;
  int wave = threadIdx.x >> 5;
  int lane = threadIdx.x & 31, lanelo = lane & 15, half = lane >> 4;
  if (threadIdx.x < 16) ssq[threadIdx.x] = 0.0f;
  __syncthreads();

  const __bf16* abase = xbf + (size_t)(mt * 16 + lanelo) * Ff + 16 * half;

  v8f acc[8];
  float sq[8];
#pragma unroll
  for (int r = 0; r < 8; r++) sq[r] = 0.0f;

#pragma unroll
  for (int i = 0; i < 8; i++) {
    int nt = wave * 8 + i;
    const __bf16* bbase = W1t + (size_t)(nt * 16 + lanelo) * Ff + 16 * half;
    v8f c = {};
#pragma unroll
    for (int ks = 0; ks < 2; ks++)
      c = wmma_bf16(load_A_at(abase, ks * 32), load_B_at(bbase, ks * 32), c);
    float bias = b1[nt * 16 + lanelo];
#pragma unroll
    for (int r = 0; r < 8; r++) { c[r] += bias; sq[r] += c[r] * c[r]; }
    acc[i] = c;
  }
#pragma unroll
  for (int r = 0; r < 8; r++) atomicAdd(&ssq[r + 8 * half], sq[r]);
  __syncthreads();

  float scale[8];
#pragma unroll
  for (int r = 0; r < 8; r++) {
    float nrm = sqrtf(ssq[r + 8 * half]);
    scale[r] = 22.62741699796952f / fmaxf(nrm, 1e-12f);  // sqrt(512)/max(||h||,eps)
  }
#pragma unroll
  for (int i = 0; i < 8; i++) {
    int nt = wave * 8 + i;
#pragma unroll
    for (int r = 0; r < 8; r++) {
      int row = mt * 16 + r + 8 * half;
      h0[(size_t)row * Hh + nt * 16 + lanelo] = (__bf16)(acc[i][r] * scale[r]);
    }
  }
}

// ---- xg = act @ Wx + b : [BT,512] x [512,2048] -> fp32 [BT,2048], 2x2 register blocking ----
// grid = (BT/32, 16); block = 128; wave -> 2 M-tiles x 2 N-tiles, K=512 (16 ksteps, 4 WMMA each).
__global__ void k_xg(const __bf16* __restrict__ act, const __bf16* __restrict__ Wt,
                     const float* __restrict__ bias, float* __restrict__ xg) {
  int mt0 = blockIdx.x * 2;
  int wave = threadIdx.x >> 5;
  int nt0 = (blockIdx.y * 4 + wave) * 2;
  int lane = threadIdx.x & 31, lanelo = lane & 15, half = lane >> 4;

  const __bf16* abase0 = act + (size_t)(mt0 * 16 + lanelo) * Hh + 16 * half;
  const __bf16* abase1 = abase0 + (size_t)16 * Hh;
  const __bf16* bbase0 = Wt + (size_t)(nt0 * 16 + lanelo) * Hh + 16 * half;
  const __bf16* bbase1 = bbase0 + (size_t)16 * Hh;

  v8f acc[2][2] = {};
#pragma unroll
  for (int ks = 0; ks < 16; ks++) {
    int k0 = ks * 32;
    v16bf a0 = load_A_at(abase0, k0);
    v16bf a1 = load_A_at(abase1, k0);
    v16bf b0 = load_B_at(bbase0, k0);
    v16bf b1 = load_B_at(bbase1, k0);
    acc[0][0] = wmma_bf16(a0, b0, acc[0][0]);
    acc[0][1] = wmma_bf16(a0, b1, acc[0][1]);
    acc[1][0] = wmma_bf16(a1, b0, acc[1][0]);
    acc[1][1] = wmma_bf16(a1, b1, acc[1][1]);
  }
#pragma unroll
  for (int ni = 0; ni < 2; ni++) {
    float bv = bias[(nt0 + ni) * 16 + lanelo];
#pragma unroll
    for (int mi = 0; mi < 2; mi++) {
#pragma unroll
      for (int r = 0; r < 8; r++)
        xg[(size_t)((mt0 + mi) * 16 + r + 8 * half) * H4 + (nt0 + ni) * 16 + lanelo] =
            acc[mi][ni][r] + bv;
    }
  }
}

// ---- one LSTM time step: g = xg[:,t,:] + h_in @ Wh ; gate math ; h/c update ----
// grid = (B/32, 8); block = 128; wave -> 2 batch-tiles x hidden-tile jt, ALL 4 gates.
// B (Wh) tiles shared across the 2 M-tiles: 12 b128 loads per 8 WMMAs.
__global__ void k_lstm_step(const float* __restrict__ xg, const __bf16* __restrict__ Wht,
                            const __bf16* __restrict__ h_in, __bf16* __restrict__ h_out,
                            float* __restrict__ c_st, __bf16* __restrict__ seq_out,
                            float* __restrict__ h2_out, int t) {
  int mt0 = blockIdx.x * 2;
  int wave = threadIdx.x >> 5;
  int jt = blockIdx.y * 4 + wave;                 // 0..31 hidden tiles
  int lane = threadIdx.x & 31, lanelo = lane & 15, half = lane >> 4;

  const __bf16* abase0 = h_in + (size_t)(mt0 * 16 + lanelo) * Hh + 16 * half;
  const __bf16* abase1 = abase0 + (size_t)16 * Hh;
  const __bf16* bbase[4];
#pragma unroll
  for (int gg = 0; gg < 4; gg++)
    bbase[gg] = Wht + (size_t)(gg * Hh + jt * 16 + lanelo) * Hh + 16 * half;

  v8f g[2][4] = {};
#pragma unroll
  for (int ks = 0; ks < 16; ks++) {
    int k0 = ks * 32;
    v16bf a0 = load_A_at(abase0, k0);
    v16bf a1 = load_A_at(abase1, k0);
#pragma unroll
    for (int gg = 0; gg < 4; gg++) {
      v16bf b = load_B_at(bbase[gg], k0);
      g[0][gg] = wmma_bf16(a0, b, g[0][gg]);
      g[1][gg] = wmma_bf16(a1, b, g[1][gg]);
    }
  }

  int j = jt * 16 + lanelo;
#pragma unroll
  for (int mi = 0; mi < 2; mi++) {
#pragma unroll
    for (int r = 0; r < 8; r++) {
      int bb = (mt0 + mi) * 16 + r + 8 * half;
      size_t xrow = ((size_t)bb * Tt + t) * H4;
      float gi = g[mi][0][r] + xg[xrow + j];
      float gf = g[mi][1][r] + xg[xrow + Hh + j];
      float gc = g[mi][2][r] + xg[xrow + 2 * Hh + j];
      float go = g[mi][3][r] + xg[xrow + 3 * Hh + j];
      float iv = sigmoid_(gi), fv = sigmoid_(gf), ov = sigmoid_(go);
      float cb = tanhf(gc);
      int cidx = bb * Hh + j;
      float cn = fv * c_st[cidx] + iv * cb;
      float hn = ov * tanhf(cn);
      c_st[cidx] = cn;
      h_out[cidx] = (__bf16)hn;
      if (seq_out) seq_out[((size_t)bb * Tt + t) * Hh + j] = (__bf16)hn;
      if (h2_out) h2_out[cidx] = hn;
    }
  }
}

// ---- head: LayerNorm -> dense(512,256)+ReLU -> dense(256,1) ----
__global__ void k_head(const float* __restrict__ h2, const float* __restrict__ gamma,
                       const float* __restrict__ beta, const float* __restrict__ W2,
                       const float* __restrict__ b2, const float* __restrict__ W3,
                       const float* __restrict__ b3, float* __restrict__ out) {
  __shared__ float hn[512];
  __shared__ float red[256];
  int row = blockIdx.x, tid = threadIdx.x;  // 256 threads
  float x0 = h2[(size_t)row * Hh + tid];
  float x1 = h2[(size_t)row * Hh + 256 + tid];
  red[tid] = x0 + x1;
  __syncthreads();
  for (int s = 128; s > 0; s >>= 1) { if (tid < s) red[tid] += red[tid + s]; __syncthreads(); }
  float mu = red[0] / 512.0f;
  __syncthreads();
  red[tid] = (x0 - mu) * (x0 - mu) + (x1 - mu) * (x1 - mu);
  __syncthreads();
  for (int s = 128; s > 0; s >>= 1) { if (tid < s) red[tid] += red[tid + s]; __syncthreads(); }
  float rstd = rsqrtf(red[0] / 512.0f + 1e-3f);
  hn[tid]       = (x0 - mu) * rstd * gamma[tid] + beta[tid];
  hn[tid + 256] = (x1 - mu) * rstd * gamma[tid + 256] + beta[tid + 256];
  __syncthreads();
  float s = b2[tid];
  for (int k = 0; k < 512; k++) s += hn[k] * W2[k * 256 + tid];
  s = fmaxf(s, 0.0f);
  red[tid] = s * W3[tid];
  __syncthreads();
  for (int st = 128; st > 0; st >>= 1) { if (tid < st) red[tid] += red[tid + st]; __syncthreads(); }
  if (tid == 0) out[row] = red[0] + b3[0];
}

extern "C" void kernel_launch(void* const* d_in, const int* in_sizes, int n_in,
                              void* d_out, int out_size, void* d_ws, size_t ws_size,
                              hipStream_t stream) {
  (void)in_sizes; (void)n_in; (void)out_size; (void)ws_size;
  const float* x      = (const float*)d_in[0];
  const float* W1     = (const float*)d_in[1];
  const float* b1     = (const float*)d_in[2];
  const float* Wx1    = (const float*)d_in[3];
  const float* Wh1    = (const float*)d_in[4];
  const float* blstm1 = (const float*)d_in[5];
  const float* Wx2    = (const float*)d_in[6];
  const float* Wh2    = (const float*)d_in[7];
  const float* blstm2 = (const float*)d_in[8];
  const float* gamma  = (const float*)d_in[9];
  const float* beta   = (const float*)d_in[10];
  const float* W2     = (const float*)d_in[11];
  const float* b2     = (const float*)d_in[12];
  const float* W3     = (const float*)d_in[13];
  const float* b3     = (const float*)d_in[14];
  float* out = (float*)d_out;

  char* ws = (char*)d_ws;
  size_t off = 0;
  auto alloc = [&](size_t bytes) -> char* {
    char* p = ws + off;
    off += (bytes + 255) & ~(size_t)255;
    return p;
  };
  __bf16* xbf   = (__bf16*)alloc((size_t)BT * Ff * 2);
  __bf16* W1t   = (__bf16*)alloc((size_t)Hh * Ff * 2);
  __bf16* Wx1t  = (__bf16*)alloc((size_t)H4 * Hh * 2);
  __bf16* Wh1t  = (__bf16*)alloc((size_t)H4 * Hh * 2);
  __bf16* Wx2t  = (__bf16*)alloc((size_t)H4 * Hh * 2);
  __bf16* Wh2t  = (__bf16*)alloc((size_t)H4 * Hh * 2);
  __bf16* h0    = (__bf16*)alloc((size_t)BT * Hh * 2);
  __bf16* seq1  = (__bf16*)alloc((size_t)BT * Hh * 2);
  float*  xg    = (float*) alloc((size_t)BT * H4 * 4);   // reused by both LSTMs
  __bf16* h_a   = (__bf16*)alloc((size_t)Bb * Hh * 2);
  __bf16* h_b   = (__bf16*)alloc((size_t)Bb * Hh * 2);
  float*  c_st  = (float*) alloc((size_t)Bb * Hh * 4);
  float*  h2    = (float*) alloc((size_t)Bb * Hh * 4);

  // convert + transpose weights to bf16 (N-major for WMMA B operand)
  k_f32_to_bf16<<<(BT * Ff + 255) / 256, 256, 0, stream>>>(x, xbf, BT * Ff);
  k_transpose_bf16<<<(Ff * Hh + 255) / 256, 256, 0, stream>>>(W1, W1t, Ff, Hh);
  k_transpose_bf16<<<(Hh * H4 + 255) / 256, 256, 0, stream>>>(Wx1, Wx1t, Hh, H4);
  k_transpose_bf16<<<(Hh * H4 + 255) / 256, 256, 0, stream>>>(Wh1, Wh1t, Hh, H4);
  k_transpose_bf16<<<(Hh * H4 + 255) / 256, 256, 0, stream>>>(Wx2, Wx2t, Hh, H4);
  k_transpose_bf16<<<(Hh * H4 + 255) / 256, 256, 0, stream>>>(Wh2, Wh2t, Hh, H4);

  // dense1 + spectral row norm
  k_dense1_norm<<<BT / 16, 128, 0, stream>>>(xbf, W1t, b1, h0);

  // LSTM1: input projection then 256 sequential recurrent steps (double-buffered h)
  k_xg<<<dim3(BT / 32, 16), 128, 0, stream>>>(h0, Wx1t, blstm1, xg);
  k_zero_state<<<(Bb * Hh + 255) / 256, 256, 0, stream>>>(h_a, c_st, Bb * Hh);
  {
    __bf16 *hin = h_a, *hout = h_b;
    for (int t = 0; t < Tt; t++) {
      k_lstm_step<<<dim3(Bb / 32, 8), 128, 0, stream>>>(xg, Wh1t, hin, hout, c_st, seq1, nullptr, t);
      __bf16* tmp = hin; hin = hout; hout = tmp;
    }
  }

  // LSTM2
  k_xg<<<dim3(BT / 32, 16), 128, 0, stream>>>(seq1, Wx2t, blstm2, xg);
  k_zero_state<<<(Bb * Hh + 255) / 256, 256, 0, stream>>>(h_a, c_st, Bb * Hh);
  {
    __bf16 *hin = h_a, *hout = h_b;
    for (int t = 0; t < Tt; t++) {
      k_lstm_step<<<dim3(Bb / 32, 8), 128, 0, stream>>>(xg, Wh2t, hin, hout, c_st, nullptr, h2, t);
      __bf16* tmp = hin; hin = hout; hout = tmp;
    }
  }

  // LayerNorm + dense + relu + dense
  k_head<<<Bb, 256, 0, stream>>>(h2, gamma, beta, W2, b2, W3, b3, out);
}
